// OptimizedLSTM_66718021976169
// MI455X (gfx1250) — matrix operand order repaired and checked by reading, not compiled, not run on hardware
//
#include <hip/hip_runtime.h>
#include <hip/hip_bf16.h>

// ---------------------------------------------------------------------------
// Model dims (match reference)
#define BB   64
#define SS   512
#define HH   512
#define LL   2
#define NHH  8
#define HDD  64
#define G4H  2048          // 4*H
#define EPSL 1e-5f

// Use CDNA5 async global->LDS copy for the (already bf16) weight tile.
#define USE_ASYNC_B 1

typedef __attribute__((ext_vector_type(16))) __bf16 v16bf;
typedef __attribute__((ext_vector_type(8)))  __bf16 v8bf;
typedef __attribute__((ext_vector_type(8)))  float  v8f;
typedef __bf16 bf16_t;

__device__ __forceinline__ float sigmoidf_(float x) { return 1.0f / (1.0f + __expf(-x)); }

__device__ __forceinline__ float blockReduceAdd(float v, float* red) {
    int tid = threadIdx.x;
    red[tid] = v;
    __syncthreads();
    for (int s = 128; s > 0; s >>= 1) {
        if (tid < s) red[tid] += red[tid + s];
        __syncthreads();
    }
    float r = red[0];
    __syncthreads();
    return r;
}

// ---------------------------------------------------------------------------
// fp32 -> bf16 weight conversion (one-time per launch, cheap)
__global__ void f32_to_bf16_kernel(const float* __restrict__ s, bf16_t* __restrict__ d, long n) {
    long i = (long)blockIdx.x * blockDim.x + threadIdx.x;
    long stride = (long)gridDim.x * blockDim.x;
    for (; i < n; i += stride) d[i] = (bf16_t)s[i];
}

__global__ void fill_zero_kernel(float* __restrict__ p, long n) {
    long i = (long)blockIdx.x * blockDim.x + threadIdx.x;
    long stride = (long)gridDim.x * blockDim.x;
    for (; i < n; i += stride) p[i] = 0.0f;
}

// ---------------------------------------------------------------------------
// Generic WMMA GEMM:  C[M,N] = act( A[M,K] @ W[N,K]^T + bias[N] + resid[M,N] )
// A fp32 (cvt to bf16 while staging), W pre-converted bf16 (async-copied to
// LDS with global_load_async_to_lds_b128, fenced by s_wait_asynccnt).
// Block: 256 thr = 8 waves; tile 64 x TBN, BK=64, double-buffered LDS.
// Wave w: m-tile (w&3), n-range (w>>2)*(TBN/2), NT=TBN/32 n-tiles.
// Requires: M % 64 == 0, N % TBN == 0, K % 64 == 0.
template<int TBN>
__global__ __launch_bounds__(256) void gemm_wmma_bf16(
    const float* __restrict__ A, const bf16_t* __restrict__ W,
    const float* __restrict__ bias, const float* __restrict__ resid,
    float* __restrict__ C, int M, int N, int K, int relu)
{
    constexpr int TBM = 64, TBK = 64;
    constexpr int NT  = TBN / 32;                 // n-tiles per wave
    constexpr int BEPT = TBN * TBK / 256;         // B bf16 elems per thread
    constexpr int BTPR = TBK / BEPT;              // threads per B row

    __shared__ __align__(16) bf16_t As[2][TBM][TBK];
    __shared__ __align__(16) bf16_t Bs[2][TBN][TBK];

    const int tid   = threadIdx.x;
    const int lane  = tid & 31;
    const int wave  = tid >> 5;
    const int lrow  = lane & 15;
    const int lhalf = lane >> 4;
    const int m0 = blockIdx.y * TBM;
    const int n0 = blockIdx.x * TBN;
    const int wm = (wave & 3) * 16;
    const int wn = (wave >> 2) * (TBN / 2);

    const int arow = tid >> 2;                    // A staging: 4 thr/row, 16 f32 each
    const int akk  = (tid & 3) * 16;
    const int brow = tid / BTPR;                  // B staging
    const int bkk  = (tid % BTPR) * BEPT;

    v8f acc[NT];
    #pragma unroll
    for (int j = 0; j < NT; ++j)
        #pragma unroll
        for (int r = 0; r < 8; ++r) acc[j][r] = 0.0f;

    auto stage = [&](int buf, int k0) {
        // A: fp32 -> bf16 through VGPRs
        const float* ap = A + (size_t)(m0 + arow) * K + k0 + akk;
        #pragma unroll
        for (int i = 0; i < 16; ++i) As[buf][arow][akk + i] = (bf16_t)ap[i];
#if USE_ASYNC_B
        // B: pure bf16 copy -> async DMA to LDS, no VGPR round-trip
        unsigned ldsoff = (unsigned)(size_t)&Bs[buf][brow][bkk];
        unsigned long long ga =
            (unsigned long long)(W + (size_t)(n0 + brow) * K + k0 + bkk);
        #pragma unroll
        for (int i = 0; i < BEPT / 8; ++i) {
            asm volatile("global_load_async_to_lds_b128 %0, %1, off"
                         :: "v"(ldsoff + i * 16), "v"(ga + i * 16) : "memory");
        }
#else
        const bf16_t* wp = W + (size_t)(n0 + brow) * K + k0 + bkk;
        #pragma unroll
        for (int i = 0; i < BEPT / 8; ++i)
            *(v8bf*)&Bs[buf][brow][bkk + 8 * i] = *(const v8bf*)(wp + 8 * i);
#endif
    };

    stage(0, 0);
    int cur = 0;
    for (int k0 = 0; k0 < K; k0 += TBK) {
#if USE_ASYNC_B
        asm volatile("s_wait_asynccnt 0" ::: "memory");
#endif
        __syncthreads();
        if (k0 + TBK < K) stage(cur ^ 1, k0 + TBK);

        #pragma unroll
        for (int kk = 0; kk < 2; ++kk) {          // two 16x16x32 steps per stage
            v16bf a;
            {
                v8bf lo = *(const v8bf*)&As[cur][wm + lrow][kk * 32 + lhalf * 8];
                v8bf hi = *(const v8bf*)&As[cur][wm + lrow][kk * 32 + 16 + lhalf * 8];
                #pragma unroll
                for (int i = 0; i < 8; ++i) { a[i] = lo[i]; a[8 + i] = hi[i]; }
            }
            #pragma unroll
            for (int j = 0; j < NT; ++j) {
                v16bf bfr;
                v8bf x0 = *(const v8bf*)&Bs[cur][wn + j * 16 + lrow][kk * 32 + lhalf * 8];
                v8bf x1 = *(const v8bf*)&Bs[cur][wn + j * 16 + lrow][kk * 32 + 16 + lhalf * 8];
                #pragma unroll
                for (int i = 0; i < 8; ++i) { bfr[i] = x0[i]; bfr[8 + i] = x1[i]; }
                acc[j] = __builtin_amdgcn_wmma_f32_16x16x32_bf16(
                    false, a, false, bfr, (short)0, acc[j], false, false);
            }
        }
        cur ^= 1;
    }

    // ---- epilogue: bias / residual / relu ----
    #pragma unroll
    for (int j = 0; j < NT; ++j) {
        #pragma unroll
        for (int r = 0; r < 8; ++r) {
            const int gm = m0 + wm + r + 8 * lhalf;
            const int gn = n0 + wn + j * 16 + lrow;
            float v = acc[j][r] + (bias ? bias[gn] : 0.0f);
            size_t i0 = (size_t)gm * N + gn;
            if (resid) v += resid[i0];
            if (relu)  v = fmaxf(v, 0.0f);
            C[i0] = v;
        }
    }
}

// ---------------------------------------------------------------------------
// Row LayerNorm: one block per row.  out[row] = LN(in[row*inStride+inOff], D)
__global__ __launch_bounds__(256) void ln_rows_kernel(
    const float* __restrict__ in, float* __restrict__ out,
    const float* __restrict__ g, const float* __restrict__ be,
    int D, long inStride, long inOff, long outStride)
{
    __shared__ float red[256];
    const int tid = threadIdx.x;
    const float* p = in + (long)blockIdx.x * inStride + inOff;
    float* q = out + (long)blockIdx.x * outStride;
    const int npt = D >> 8;       // D/256 (D is 512 or 2048)
    float vals[8];
    float s = 0.0f, ss = 0.0f;
    for (int k = 0; k < npt; ++k) {
        float v = p[tid + (k << 8)];
        vals[k] = v; s += v; ss += v * v;
    }
    s  = blockReduceAdd(s, red);
    ss = blockReduceAdd(ss, red);
    const float mean = s / (float)D;
    const float var  = ss / (float)D - mean * mean;
    const float rstd = rsqrtf(var + EPSL);
    for (int k = 0; k < npt; ++k) {
        int idx = tid + (k << 8);
        q[idx] = (vals[k] - mean) * rstd * g[idx] + be[idx];
    }
}

// ---------------------------------------------------------------------------
// Fused LSTM cell (one block per batch row):
//  hp = LN(hpraw[b], 4H; g_hid,b_hid); z = ip[b,t] + hp; i,f,g,o = split(z)
//  c' = sig(f)*c + sig(i)*tanh(g);  h = sig(o)*tanh(LN(c', H; g_cell,b_cell))
// thread t owns columns t and t+256 with their full i/f/g/o quadruples.
__global__ __launch_bounds__(256) void lstm_step_kernel(
    const float* __restrict__ hpraw, const float* __restrict__ ip, int t,
    const float* __restrict__ ghid, const float* __restrict__ bhid,
    const float* __restrict__ gcell, const float* __restrict__ bcell,
    float* __restrict__ h, float* __restrict__ c, float* __restrict__ hseq)
{
    __shared__ float red[256];
    const int b = blockIdx.x;
    const int tid = threadIdx.x;
    const float* hp = hpraw + (size_t)b * G4H;
    const float* iprow = ip + ((size_t)b * SS + t) * G4H;

    float raw[8];
    float s = 0.0f, ss = 0.0f;
    #pragma unroll
    for (int k = 0; k < 8; ++k) {
        float v = hp[tid + (k << 8)];
        raw[k] = v; s += v; ss += v * v;
    }
    s  = blockReduceAdd(s, red);
    ss = blockReduceAdd(ss, red);
    const float mean = s / (float)G4H;
    const float rstd = rsqrtf(ss / (float)G4H - mean * mean + EPSL);

    float z[8];
    #pragma unroll
    for (int k = 0; k < 8; ++k) {
        int idx = tid + (k << 8);
        float hv = (raw[k] - mean) * rstd * ghid[idx] + bhid[idx];
        z[k] = iprow[idx] + hv;
    }

    // gate split: idx = tid+256k -> i:(k=0,1) f:(2,3) g:(4,5) o:(6,7)
    float cn[2], og[2];
    float s2 = 0.0f, ss2 = 0.0f;
    #pragma unroll
    for (int jj = 0; jj < 2; ++jj) {
        int j = tid + (jj << 8);
        float i_ = z[0 + jj], f_ = z[2 + jj], g_ = z[4 + jj], o_ = z[6 + jj];
        float cv = sigmoidf_(f_) * c[(size_t)b * HH + j] + sigmoidf_(i_) * tanhf(g_);
        c[(size_t)b * HH + j] = cv;
        cn[jj] = cv; og[jj] = sigmoidf_(o_);
        s2 += cv; ss2 += cv * cv;
    }
    s2  = blockReduceAdd(s2, red);
    ss2 = blockReduceAdd(ss2, red);
    const float mean2 = s2 / (float)HH;
    const float rstd2 = rsqrtf(ss2 / (float)HH - mean2 * mean2 + EPSL);

    #pragma unroll
    for (int jj = 0; jj < 2; ++jj) {
        int j = tid + (jj << 8);
        float lnc = (cn[jj] - mean2) * rstd2 * gcell[j] + bcell[j];
        float hv = og[jj] * tanhf(lnc);
        h[(size_t)b * HH + j] = hv;
        hseq[((size_t)b * SS + t) * HH + j] = hv;
    }
}

// ---------------------------------------------------------------------------
// Highway elementwise: out = sig(tg)*relu(tr) + sig(cg)*hs (+ skip)
__global__ __launch_bounds__(256) void highway_ew_kernel(
    const float* __restrict__ tg, const float* __restrict__ cg,
    const float* __restrict__ tr, const float* __restrict__ hs,
    const float* __restrict__ skip, float* __restrict__ out, long n)
{
    long i = (long)blockIdx.x * blockDim.x + threadIdx.x;
    long stride = (long)gridDim.x * blockDim.x;
    for (; i < n; i += stride) {
        float v = sigmoidf_(tg[i]) * fmaxf(tr[i], 0.0f) + sigmoidf_(cg[i]) * hs[i];
        if (skip) v += skip[i];
        out[i] = v;
    }
}

// ---------------------------------------------------------------------------
// Flash attention (WMMA, online softmax).  grid = (S/128, NH, B), 256 thr.
// Each wave owns a private 16-row q strip -> softmax row stats stay in-wave
// (row of a 16x16 C tile spans 16 lanes -> 4 shfl_xor hops).
__global__ __launch_bounds__(256) void flash_attn_kernel(
    const float* __restrict__ Q, const float* __restrict__ Kg,
    const float* __restrict__ Vg, float* __restrict__ O)
{
    __shared__ __align__(16) bf16_t Qs[128][64];
    __shared__ __align__(16) bf16_t Ks[64][64];
    __shared__ __align__(16) bf16_t Vs[64][80];   // transposed [hd][key], padded
    __shared__ __align__(16) bf16_t Ps[128][64];

    const int tid   = threadIdx.x;
    const int lane  = tid & 31;
    const int wave  = tid >> 5;
    const int lrow  = lane & 15;
    const int lhalf = lane >> 4;
    const int qt    = blockIdx.x;
    const int head  = blockIdx.y;
    const int b     = blockIdx.z;
    const int wm    = wave * 16;
    const float scale = 0.125f;   // 1/sqrt(64)

    // stage Q (128 x 64)
    {
        int r = tid >> 1, cc = (tid & 1) * 32;
        const float* qp = Q + ((size_t)(b * SS + qt * 128 + r)) * HH + head * HDD + cc;
        #pragma unroll
        for (int i = 0; i < 32; ++i) Qs[r][cc + i] = (bf16_t)qp[i];
    }

    v8f o[4];
    float mstate[8], lstate[8];
    #pragma unroll
    for (int j = 0; j < 4; ++j)
        #pragma unroll
        for (int r = 0; r < 8; ++r) o[j][r] = 0.0f;
    #pragma unroll
    for (int r = 0; r < 8; ++r) { mstate[r] = -1e30f; lstate[r] = 0.0f; }

    for (int kt = 0; kt < SS / 64; ++kt) {
        __syncthreads();
        // stage K [key][hd] and V transposed [hd][key]
        {
            int r = tid >> 2, cc = (tid & 3) * 16;
            const float* kp = Kg + ((size_t)(b * SS + kt * 64 + r)) * HH + head * HDD + cc;
            const float* vp = Vg + ((size_t)(b * SS + kt * 64 + r)) * HH + head * HDD + cc;
            #pragma unroll
            for (int i = 0; i < 16; ++i) {
                Ks[r][cc + i] = (bf16_t)kp[i];
                Vs[cc + i][r] = (bf16_t)vp[i];
            }
        }
        __syncthreads();

        // S = Q @ K^T  (16x64 per wave, K-dim 64 -> 2 wmma steps x 4 n-tiles)
        v8f sf[4];
        #pragma unroll
        for (int j = 0; j < 4; ++j)
            #pragma unroll
            for (int r = 0; r < 8; ++r) sf[j][r] = 0.0f;

        #pragma unroll
        for (int kk = 0; kk < 2; ++kk) {
            v16bf a;
            {
                v8bf lo = *(const v8bf*)&Qs[wm + lrow][kk * 32 + lhalf * 8];
                v8bf hi = *(const v8bf*)&Qs[wm + lrow][kk * 32 + 16 + lhalf * 8];
                #pragma unroll
                for (int i = 0; i < 8; ++i) { a[i] = lo[i]; a[8 + i] = hi[i]; }
            }
            #pragma unroll
            for (int j = 0; j < 4; ++j) {
                v16bf bfr;
                v8bf x0 = *(const v8bf*)&Ks[j * 16 + lrow][kk * 32 + lhalf * 8];
                v8bf x1 = *(const v8bf*)&Ks[j * 16 + lrow][kk * 32 + 16 + lhalf * 8];
                #pragma unroll
                for (int i = 0; i < 8; ++i) { bfr[i] = x0[i]; bfr[8 + i] = x1[i]; }
                sf[j] = __builtin_amdgcn_wmma_f32_16x16x32_bf16(false, a, false, bfr, (short)0, sf[j], false, false);
            }
        }

        // online softmax per row (row spans 16 lanes within a half)
        #pragma unroll
        for (int r = 0; r < 8; ++r) {
            float tm = -1e30f;
            #pragma unroll
            for (int j = 0; j < 4; ++j) { sf[j][r] *= scale; tm = fmaxf(tm, sf[j][r]); }
            #pragma unroll
            for (int off = 1; off < 16; off <<= 1) tm = fmaxf(tm, __shfl_xor(tm, off, 32));
            float mnew = fmaxf(mstate[r], tm);
            float corr = __expf(mstate[r] - mnew);
            float psum = 0.0f;
            #pragma unroll
            for (int j = 0; j < 4; ++j) {
                float p = __expf(sf[j][r] - mnew);
                sf[j][r] = p; psum += p;
            }
            #pragma unroll
            for (int off = 1; off < 16; off <<= 1) psum += __shfl_xor(psum, off, 32);
            lstate[r] = lstate[r] * corr + psum;
            mstate[r] = mnew;
            #pragma unroll
            for (int j = 0; j < 4; ++j) o[j][r] *= corr;
        }

        // spill P (bf16) to this wave's private LDS strip, reload as A-frags
        #pragma unroll
        for (int j = 0; j < 4; ++j)
            #pragma unroll
            for (int r = 0; r < 8; ++r)
                Ps[wm + r + 8 * lhalf][j * 16 + lrow] = (bf16_t)sf[j][r];

        // O += P @ V   (B-frag from transposed V tile)
        #pragma unroll
        for (int kk = 0; kk < 2; ++kk) {
            v16bf a;
            {
                v8bf lo = *(const v8bf*)&Ps[wm + lrow][kk * 32 + lhalf * 8];
                v8bf hi = *(const v8bf*)&Ps[wm + lrow][kk * 32 + 16 + lhalf * 8];
                #pragma unroll
                for (int i = 0; i < 8; ++i) { a[i] = lo[i]; a[8 + i] = hi[i]; }
            }
            #pragma unroll
            for (int j = 0; j < 4; ++j) {
                v16bf bfr;
                v8bf x0 = *(const v8bf*)&Vs[j * 16 + lrow][kk * 32 + lhalf * 8];
                v8bf x1 = *(const v8bf*)&Vs[j * 16 + lrow][kk * 32 + 16 + lhalf * 8];
                #pragma unroll
                for (int i = 0; i < 8; ++i) { bfr[i] = x0[i]; bfr[8 + i] = x1[i]; }
                o[j] = __builtin_amdgcn_wmma_f32_16x16x32_bf16(false, a, false, bfr, (short)0, o[j], false, false);
            }
        }
    }

    // write ctx = O / l
    #pragma unroll
    for (int j = 0; j < 4; ++j)
        #pragma unroll
        for (int r = 0; r < 8; ++r) {
            int qrow = qt * 128 + wm + r + 8 * lhalf;
            int hd   = j * 16 + lrow;
            O[((size_t)(b * SS + qrow)) * HH + head * HDD + hd] = o[j][r] / lstate[r];
        }
}

// ---------------------------------------------------------------------------
// Final head: out[b] = bp2 + mid[b,:] . Wp2[:]
__global__ void head_out_kernel(const float* __restrict__ mid,
                                const float* __restrict__ Wp2,
                                const float* __restrict__ bp2,
                                float* __restrict__ out)
{
    int b = threadIdx.x;
    if (b < BB) {
        float s = bp2[0];
        for (int j = 0; j < HH / 2; ++j) s += mid[b * (HH / 2) + j] * Wp2[j];
        out[b] = s;
    }
}

// ---------------------------------------------------------------------------
extern "C" void kernel_launch(void* const* d_in, const int* in_sizes, int n_in,
                              void* d_out, int out_size, void* d_ws, size_t ws_size,
                              hipStream_t stream) {
    (void)in_sizes; (void)n_in; (void)out_size; (void)ws_size;
    // inputs (setup_inputs order)
    const float* x      = (const float*)d_in[0];
    const float* Wi     = (const float*)d_in[1];
    const float* bi     = (const float*)d_in[2];
    const float* Wh     = (const float*)d_in[3];
    const float* bh     = (const float*)d_in[4];
    const float* g_in   = (const float*)d_in[5];
    const float* b_in   = (const float*)d_in[6];
    const float* g_hid  = (const float*)d_in[7];
    const float* b_hid  = (const float*)d_in[8];
    const float* g_cell = (const float*)d_in[9];
    const float* b_cell = (const float*)d_in[10];
    const float* hw_Wt  = (const float*)d_in[11];
    const float* hw_bt  = (const float*)d_in[12];
    const float* hw_Wc  = (const float*)d_in[13];
    const float* hw_bc  = (const float*)d_in[14];
    const float* hw_Wh  = (const float*)d_in[15];
    const float* hw_bh  = (const float*)d_in[16];
    const float* Wq = (const float*)d_in[17]; const float* bq = (const float*)d_in[18];
    const float* Wk = (const float*)d_in[19]; const float* bk = (const float*)d_in[20];
    const float* Wv = (const float*)d_in[21]; const float* bv = (const float*)d_in[22];
    const float* Wo = (const float*)d_in[23]; const float* bo = (const float*)d_in[24];
    const float* att_g = (const float*)d_in[25]; const float* att_b = (const float*)d_in[26];
    const float* on_g  = (const float*)d_in[27]; const float* on_b  = (const float*)d_in[28];
    const float* Wp1 = (const float*)d_in[29]; const float* bp1 = (const float*)d_in[30];
    const float* Wp2 = (const float*)d_in[31]; const float* bp2 = (const float*)d_in[32];
    float* out = (float*)d_out;

    // ---- workspace layout ----
    char* base = (char*)d_ws;
    size_t off = 0;
    auto alloc = [&](size_t bytes) -> char* {
        char* p = base + off;
        off = (off + bytes + 255) & ~(size_t)255;
        return p;
    };
    bf16_t* WiB  = (bf16_t*)alloc((size_t)LL * G4H * HH * 2);
    bf16_t* WhB  = (bf16_t*)alloc((size_t)LL * G4H * HH * 2);
    bf16_t* hwTB = (bf16_t*)alloc((size_t)LL * HH * HH * 2);
    bf16_t* hwCB = (bf16_t*)alloc((size_t)LL * HH * HH * 2);
    bf16_t* hwHB = (bf16_t*)alloc((size_t)LL * HH * HH * 2);
    bf16_t* WqB  = (bf16_t*)alloc((size_t)HH * HH * 2);
    bf16_t* WkB  = (bf16_t*)alloc((size_t)HH * HH * 2);
    bf16_t* WvB  = (bf16_t*)alloc((size_t)HH * HH * 2);
    bf16_t* WoB  = (bf16_t*)alloc((size_t)HH * HH * 2);
    bf16_t* Wp1B = (bf16_t*)alloc((size_t)(HH / 2) * HH * 2);
    const size_t SEQN = (size_t)BB * SS * HH;           // 16,777,216 floats
    float* big    = (float*)alloc((size_t)BB * SS * G4H * 4);   // 256MB, time-shared
    float* seqws  = (float*)alloc(SEQN * 4);
    float* hseqb  = (float*)alloc(SEQN * 4);
    float* hstate = (float*)alloc((size_t)BB * HH * 4);
    float* cstate = (float*)alloc((size_t)BB * HH * 4);
    float* hpraw  = (float*)alloc((size_t)BB * G4H * 4);
    float* lastln = (float*)alloc((size_t)BB * HH * 4);
    float* mid    = (float*)alloc((size_t)BB * (HH / 2) * 4);

    float* ipbuf = big;                 // [B*S, 4H] (LSTM phase)
    float* hwT = big;                   // highway raw (reuse after recurrence)
    float* hwC = big + SEQN;
    float* hwH = big + 2 * SEQN;
    float* qb  = big;                   // attention phase reuse
    float* kb  = big + SEQN;
    float* vb  = big + 2 * SEQN;
    float* ctx = big + 3 * SEQN;

    // ---- one-time weight conversion fp32 -> bf16 ----
    auto conv = [&](const float* s, bf16_t* d, long n) {
        f32_to_bf16_kernel<<<1024, 256, 0, stream>>>(s, d, n);
    };
    conv(Wi, WiB, (long)LL * G4H * HH);
    conv(Wh, WhB, (long)LL * G4H * HH);
    conv(hw_Wt, hwTB, (long)LL * HH * HH);
    conv(hw_Wc, hwCB, (long)LL * HH * HH);
    conv(hw_Wh, hwHB, (long)LL * HH * HH);
    conv(Wq, WqB, (long)HH * HH);
    conv(Wk, WkB, (long)HH * HH);
    conv(Wv, WvB, (long)HH * HH);
    conv(Wo, WoB, (long)HH * HH);
    conv(Wp1, Wp1B, (long)(HH / 2) * HH);

    const int MBS = BB * SS;            // 32768
    // ---- LSTM stack ----
    for (int l = 0; l < LL; ++l) {
        const float* seqIn = (l == 0) ? x : seqws;
        // ip = seq @ Wi^T + bi  -> LN(g_in, b_in)
        gemm_wmma_bf16<128><<<dim3(G4H / 128, MBS / 64), 256, 0, stream>>>(
            seqIn, WiB + (size_t)l * G4H * HH, bi + l * G4H, nullptr,
            ipbuf, MBS, G4H, HH, 0);
        ln_rows_kernel<<<MBS, 256, 0, stream>>>(
            ipbuf, ipbuf, g_in + l * G4H, b_in + l * G4H, G4H, G4H, 0, G4H);
        // zero recurrent state
        fill_zero_kernel<<<64, 256, 0, stream>>>(hstate, (long)BB * HH);
        fill_zero_kernel<<<64, 256, 0, stream>>>(cstate, (long)BB * HH);
        // recurrence (small-M GEMM uses TBN=64 for 2x the workgroup parallelism)
        for (int t = 0; t < SS; ++t) {
            gemm_wmma_bf16<64><<<dim3(G4H / 64, 1), 256, 0, stream>>>(
                hstate, WhB + (size_t)l * G4H * HH, bh + l * G4H, nullptr,
                hpraw, BB, G4H, HH, 0);
            lstm_step_kernel<<<BB, 256, 0, stream>>>(
                hpraw, ipbuf, t, g_hid + l * G4H, b_hid + l * G4H,
                g_cell + l * HH, b_cell + l * HH, hstate, cstate, hseqb);
        }
        // highway: three GEMMs on h_seq, then fused gate elementwise
        gemm_wmma_bf16<128><<<dim3(HH / 128, MBS / 64), 256, 0, stream>>>(
            hseqb, hwTB + (size_t)l * HH * HH, hw_bt + l * HH, nullptr, hwT, MBS, HH, HH, 0);
        gemm_wmma_bf16<128><<<dim3(HH / 128, MBS / 64), 256, 0, stream>>>(
            hseqb, hwCB + (size_t)l * HH * HH, hw_bc + l * HH, nullptr, hwC, MBS, HH, HH, 0);
        gemm_wmma_bf16<128><<<dim3(HH / 128, MBS / 64), 256, 0, stream>>>(
            hseqb, hwHB + (size_t)l * HH * HH, hw_bh + l * HH, nullptr, hwH, MBS, HH, HH, 0);
        const float* skip = (l > 0 && ((l + 1) % 2 == 0)) ? seqIn : nullptr;
        highway_ew_kernel<<<4096, 256, 0, stream>>>(hwT, hwC, hwH, hseqb, skip, seqws, (long)SEQN);
    }

    // ---- attention ----
    gemm_wmma_bf16<128><<<dim3(HH / 128, MBS / 64), 256, 0, stream>>>(
        seqws, WqB, bq, nullptr, qb, MBS, HH, HH, 0);
    gemm_wmma_bf16<128><<<dim3(HH / 128, MBS / 64), 256, 0, stream>>>(
        seqws, WkB, bk, nullptr, kb, MBS, HH, HH, 0);
    gemm_wmma_bf16<128><<<dim3(HH / 128, MBS / 64), 256, 0, stream>>>(
        seqws, WvB, bv, nullptr, vb, MBS, HH, HH, 0);
    flash_attn_kernel<<<dim3(SS / 128, NHH, BB), 256, 0, stream>>>(qb, kb, vb, ctx);
    // att = LN(ctx @ Wo^T + bo + seq)
    gemm_wmma_bf16<128><<<dim3(HH / 128, MBS / 64), 256, 0, stream>>>(
        ctx, WoB, bo, seqws, hseqb, MBS, HH, HH, 0);
    ln_rows_kernel<<<MBS, 256, 0, stream>>>(hseqb, hseqb, att_g, att_b, HH, HH, 0, HH);

    // ---- head: last timestep -> output LN -> MLP ----
    ln_rows_kernel<<<BB, 256, 0, stream>>>(
        hseqb, lastln, on_g, on_b, HH, (long)SS * HH, (long)(SS - 1) * HH, HH);
    gemm_wmma_bf16<128><<<dim3((HH / 2) / 128, 1), 256, 0, stream>>>(
        lastln, Wp1B, bp1, nullptr, mid, BB, HH / 2, HH, 1);
    head_out_kernel<<<1, 64, 0, stream>>>(mid, Wp2, bp2, out);
}
